// UpsampleBlock_84859963834750
// MI455X (gfx1250) — compile-verified, matrix operand order checked by reading, not run on hardware
//
#include <hip/hip_runtime.h>
#include <hip/hip_bf16.h>

// ---------------------------------------------------------------------------
// MI455X (gfx1250) implementation of UpsampleBlock forward.
// Heavy conv stages run as implicit-GEMM on the WMMA pipe (f16 in, f32 acc)
// with CDNA5 async global->LDS copies (ASYNCcnt) in a double-buffered,
// single-barrier pipeline. Everything else (BN stats, attention scalars,
// resize, Chan-Vese) is bandwidth-trivial VALU work.
// ---------------------------------------------------------------------------

typedef __attribute__((ext_vector_type(16))) _Float16 v16h;
typedef __attribute__((ext_vector_type(8)))  _Float16 v8h;
typedef __attribute__((ext_vector_type(8)))  float    v8f;

#define NB   8
#define NH   128
#define NW   128
#define MTOT (NB*NH*NW)          // 131072 pixels
#define PI_F 3.14159265358979f

// ---- CDNA5 async copy helpers (ISA 15.18.3 opc 98, SOPP opc 74) ----------
__device__ __forceinline__ void async_b128(const void* lds_dst, const void* gsrc)
{
    const unsigned l = (unsigned)(unsigned long long)(uintptr_t)lds_dst; // LDS offset
    const unsigned long long g = (unsigned long long)(uintptr_t)gsrc;
    asm volatile("global_load_async_to_lds_b128 %0, %1, off"
                 :: "v"(l), "v"(g) : "memory");
}
__device__ __forceinline__ void wait_async0()
{
    asm volatile("s_wait_asynccnt 0x0" ::: "memory");
}

// ========================= WMMA implicit-GEMM ==============================
// A: activations f16, row-major [pixel][lda], K channels used per tap.
// Wt: weights f16, [tap][Cout][K].
// Workgroup: 256 threads = 8 wave32s, tile 128(M = one image row) x 64(N).
// Each wave: 32x32 tile = 2x2 WMMA 16x16 frags, K consumed 32 at a time.
// Double-buffered LDS fed by global_load_async_to_lds_b128.
__global__ __launch_bounds__(256)
void k_gemm_wmma(const _Float16* __restrict__ A, int lda,
                 const _Float16* __restrict__ Wt,
                 const float* __restrict__ bias,
                 float* __restrict__ Out,
                 int K, int Cout, int taps,
                 int mode, int ty, int tx)   // mode 0: [p][Cout]; 1: deconv scatter
{
    __shared__ _Float16 As[2][128][40];   // 128 x 32 halves, stride 40 (16B aligned)
    __shared__ _Float16 Bs[2][64][40];

    const int tid   = threadIdx.x;
    const int bb    = blockIdx.y / NH;        // batch
    const int hh    = blockIdx.y % NH;        // output row
    const int nbase = blockIdx.x * 64;

    const int wave = tid >> 5;                // wave32
    const int lane = tid & 31;
    const int m0   = (wave >> 1) * 32;
    const int n0   = (wave & 1) * 32;
    const int lr   = lane & 15;               // frag row / col
    const int lk   = (lane >> 4) * 8;         // frag K base (0 or 8)

    const int ar   = tid >> 1;                // A tile row (0..127)
    const int aseg = tid & 1;                 // halves [aseg*16, +16)
    const int bnr  = tid & 63;                // B tile row (0..63)
    const int bseg = tid >> 6;                // halves [bseg*8, +8)

    // valid dy range for SAME padding (uniform per block)
    int dylo = 0, ntap = 1;
    if (taps == 9) {
        dylo = (hh == 0) ? 0 : -1;
        const int dyhi = (hh == NH - 1) ? 0 : 1;
        ntap = (dyhi - dylo + 1) * 3;
    }
    const int kcPerTap = K >> 5;
    const int nch = ntap * kcPerTap;          // number of 32-wide K chunks

    v8f zero = {};
    v8f acc[2][2];
    for (int i = 0; i < 2; ++i)
        for (int j = 0; j < 2; ++j) acc[i][j] = zero;

    // issue async copies for chunk c into LDS buffer `buf`
    auto issue = [&](int c, int buf) {
        const int ti = c / kcPerTap;
        const int kc = (c - ti * kcPerTap) << 5;
        int dy = 0, dx = 0, t = 0;
        if (taps == 9) {
            dy = dylo + ti / 3;
            dx = ti % 3 - 1;
            t  = (dy + 1) * 3 + (dx + 1);     // absolute tap for weight addressing
        }
        const int hp = hh + dy;               // always in range by construction
        const size_t prow = (size_t)(bb * NH + hp) * NW;
        // B tile: 16B per thread
        {
            const _Float16* src =
                Wt + ((size_t)(t * Cout + nbase + bnr)) * K + kc + bseg * 8;
            async_b128(&Bs[buf][bnr][bseg * 8], src);
        }
        // A tile: 32B per thread (2 x b128); zero-fill SAME pad columns
        {
            const int wp = ar + dx;
            if (wp >= 0 && wp < NW) {
                const _Float16* src = A + (prow + (size_t)wp) * lda + kc + aseg * 16;
                async_b128(&As[buf][ar][aseg * 16],     src);
                async_b128(&As[buf][ar][aseg * 16 + 8], src + 8);
            } else {
                const uint4 z = make_uint4(0u, 0u, 0u, 0u);
                *(uint4*)&As[buf][ar][aseg * 16]     = z;
                *(uint4*)&As[buf][ar][aseg * 16 + 8] = z;
            }
        }
    };

    issue(0, 0);
    for (int c = 0; c < nch; ++c) {
        const int buf = c & 1;
        wait_async0();          // my chunk-c copies have landed in LDS
        __syncthreads();        // everyone's have (and buf^1 readers are done)
        if (c + 1 < nch) issue(c + 1, buf ^ 1);   // overlap with compute

        // ---- fragments per ISA 7.12.2 layout ----
        union FA { v16h f; v8h h[2]; };
        FA a[2], b[2];
        for (int i = 0; i < 2; ++i) {
            const int m = m0 + i * 16 + lr;
            a[i].h[0] = *(const v8h*)&As[buf][m][lk];
            a[i].h[1] = *(const v8h*)&As[buf][m][lk + 16];
        }
        for (int j = 0; j < 2; ++j) {
            const int n = n0 + j * 16 + lr;
            b[j].h[0] = *(const v8h*)&Bs[buf][n][lk];
            b[j].h[1] = *(const v8h*)&Bs[buf][n][lk + 16];
        }
        for (int i = 0; i < 2; ++i)
            for (int j = 0; j < 2; ++j)
                acc[i][j] = __builtin_amdgcn_wmma_f32_16x16x32_f16(
                    false, a[i].f, false, b[j].f,
                    (short)0, acc[i][j], false, false);
    }

    // ---- epilogue: +bias, f32 out (mode 1 scatters for the deconv) ----
    const int mr = (lane >> 4) * 8;
    for (int i = 0; i < 2; ++i) {
        for (int j = 0; j < 2; ++j) {
            const int ng = nbase + n0 + j * 16 + lr;
            const float bv = bias[ng];
            for (int r = 0; r < 8; ++r) {
                const int m = m0 + i * 16 + mr + r;
                const int p = blockIdx.y * NW + m;      // pixel index
                const float v = acc[i][j][r] + bv;
                if (mode == 0) {
                    Out[(size_t)p * Cout + ng] = v;
                } else {
                    const int pb = p >> 14;
                    const int ph = (p >> 7) & 127;
                    const int pw = p & 127;
                    const int oh = 2 * ph + ty, ow = 2 * pw + tx;
                    Out[(((size_t)(pb * 256 + oh)) * 256 + ow) * 128 + ng] = v;
                }
            }
        }
    }
}

// ========================= small helper kernels ============================

__global__ void k_concat_cvt(const float* __restrict__ in, const float* __restrict__ res,
                             _Float16* __restrict__ x)
{
    size_t idx = (size_t)blockIdx.x * 256 + threadIdx.x;
    if (idx >= (size_t)MTOT * 256) return;
    const int c = (int)(idx & 255);
    const size_t m = idx >> 8;
    const float v = (c < 128) ? in[m * 128 + c] : res[m * 128 + (c - 128)];
    x[idx] = (_Float16)v;
}

// (kh,kw,Cin,Cout) f32  ->  [tap][Cout][Cin] f16
__global__ void k_wt_cvt(const float* __restrict__ w, _Float16* __restrict__ wt,
                         int taps, int Cin, int Cout)
{
    int idx = blockIdx.x * 256 + threadIdx.x;
    const int total = taps * Cin * Cout;
    if (idx >= total) return;
    const int t  = idx / (Cin * Cout);
    const int r  = idx % (Cin * Cout);
    const int co = r / Cin;
    const int ci = r % Cin;
    wt[idx] = (_Float16)w[(size_t)(t * Cin + ci) * Cout + co];
}

// per-channel batch-norm statistics: mean + rsqrt(var+eps)
__global__ void k_stats(const float* __restrict__ in, int Mtot, int C,
                        float* __restrict__ mean, float* __restrict__ inv)
{
    const int c = blockIdx.x;
    __shared__ float s1[256], s2[256];
    float a = 0.f, b = 0.f;
    for (int m = threadIdx.x; m < Mtot; m += 256) {
        const float v = in[(size_t)m * C + c];
        a += v; b += v * v;
    }
    s1[threadIdx.x] = a; s2[threadIdx.x] = b;
    __syncthreads();
    for (int st = 128; st > 0; st >>= 1) {
        if (threadIdx.x < st) {
            s1[threadIdx.x] += s1[threadIdx.x + st];
            s2[threadIdx.x] += s2[threadIdx.x + st];
        }
        __syncthreads();
    }
    if (threadIdx.x == 0) {
        const float mu = s1[0] / (float)Mtot;
        const float var = s2[0] / (float)Mtot - mu * mu;
        mean[c] = mu;
        inv[c]  = rsqrtf(var + 1e-5f);
    }
}

__global__ void k_bn_apply(const float* __restrict__ in, int Mtot, int C,
                           const float* __restrict__ mean, const float* __restrict__ inv,
                           const float* __restrict__ g, const float* __restrict__ be,
                           int relu, int sig,
                           float* __restrict__ f32out, _Float16* __restrict__ f16out)
{
    size_t idx = (size_t)blockIdx.x * 256 + threadIdx.x;
    const size_t total = (size_t)Mtot * C;
    if (idx >= total) return;
    const int c = (int)(idx % C);
    float v = g[c] * (in[idx] - mean[c]) * inv[c] + be[c];
    if (relu) v = fmaxf(v, 0.f);
    if (sig)  v = 1.f / (1.f + expf(-v));
    if (f32out) f32out[idx] = v;
    if (f16out) f16out[idx] = (_Float16)v;
}

// relu( BN(a1raw) + BN(a2raw) )
__global__ void k_add_bn_relu(const float* __restrict__ a1, const float* __restrict__ a2,
                              const float* __restrict__ m0, const float* __restrict__ i0,
                              const float* __restrict__ g0, const float* __restrict__ be0,
                              const float* __restrict__ m1, const float* __restrict__ i1,
                              const float* __restrict__ g1, const float* __restrict__ be1,
                              float* __restrict__ out)
{
    size_t idx = (size_t)blockIdx.x * 256 + threadIdx.x;
    if (idx >= (size_t)MTOT * 128) return;
    const int c = (int)(idx & 127);
    const float v = (g0[c] * (a1[idx] - m0[c]) * i0[c] + be0[c]) +
                    (g1[c] * (a2[idx] - m1[c]) * i1[c] + be1[c]);
    out[idx] = fmaxf(v, 0.f);
}

// 1x1 conv with Cout=1: per-pixel dot of 128 channels
__global__ void k_dot128(const float* __restrict__ in, const float* __restrict__ w,
                         const float* __restrict__ bias, float* __restrict__ out)
{
    __shared__ float ws[128];
    if (threadIdx.x < 128) ws[threadIdx.x] = w[threadIdx.x];
    __syncthreads();
    const int m = blockIdx.x * 256 + threadIdx.x;
    if (m >= MTOT) return;
    const float* row = in + (size_t)m * 128;
    float s = bias[0];
#pragma unroll 8
    for (int k = 0; k < 128; ++k) s += row[k] * ws[k];
    out[m] = s;
}

// bilinear 512->128 downsample + sigmoid
__global__ void k_resize_sigmoid(const float* __restrict__ img, float* __restrict__ out)
{
    const int idx = blockIdx.x * 256 + threadIdx.x;
    if (idx >= MTOT) return;
    const int b = idx >> 14, r = idx & 16383;
    const int i = r >> 7, j = r & 127;
    const float sy = (i + 0.5f) * 4.f - 0.5f;
    const float sx = (j + 0.5f) * 4.f - 0.5f;
    int y0 = (int)floorf(sy), x0 = (int)floorf(sx);
    const float fy = sy - (float)y0, fx = sx - (float)x0;
    const int y1 = min(y0 + 1, 511), x1 = min(x0 + 1, 511);
    y0 = max(y0, 0); x0 = max(x0, 0);
    const float* ib = img + (size_t)b * 512 * 512;
    const float v00 = ib[y0 * 512 + x0], v01 = ib[y0 * 512 + x1];
    const float v10 = ib[y1 * 512 + x0], v11 = ib[y1 * 512 + x1];
    const float v = v00 * (1.f - fy) * (1.f - fx) + v01 * (1.f - fy) * fx +
                    v10 * fy * (1.f - fx) + v11 * fy * fx;
    out[idx] = 1.f / (1.f + expf(-v));
}

// t4raw = w4*(imr + BN3(t3raw)) + b4
__global__ void k_t4(const float* __restrict__ t3raw, const float* __restrict__ imr,
                     const float* __restrict__ m3, const float* __restrict__ i3,
                     const float* __restrict__ g3, const float* __restrict__ be3,
                     const float* __restrict__ w4, const float* __restrict__ b4,
                     float* __restrict__ out)
{
    const int idx = blockIdx.x * 256 + threadIdx.x;
    if (idx >= MTOT) return;
    const float t3 = g3[0] * (t3raw[idx] - m3[0]) * i3[0] + be3[0];
    out[idx] = w4[0] * (imr[idx] + t3) + b4[0];
}

// 5x5 Gaussian of (2a-1), SAME zero padding -> edt output and phi0
__global__ void k_conv_phi(const float* __restrict__ a, float* __restrict__ edt_out,
                           float* __restrict__ phi0)
{
    const int idx = blockIdx.x * 256 + threadIdx.x;
    if (idx >= MTOT) return;
    const int b = idx >> 14, r = idx & 16383;
    const int i = r >> 7, j = r & 127;
    const float w1[5] = {0.41111229f, 0.80073740f, 1.0f, 0.80073740f, 0.41111229f};
    const float inv = 1.0f / 3.42369938f;
    float s = 0.f;
    for (int dy = -2; dy <= 2; ++dy) {
        const int y = i + dy; if (y < 0 || y > 127) continue;
        for (int dx = -2; dx <= 2; ++dx) {
            const int x = j + dx; if (x < 0 || x > 127) continue;
            const float v = 2.f * a[(b << 14) + (y << 7) + x] - 1.f;
            s += v * w1[dy + 2] * w1[dx + 2];
        }
    }
    s *= inv * inv;
    edt_out[idx] = s;
    phi0[idx] = s;
}

// ----------------------------- Chan-Vese -----------------------------------
__device__ __forceinline__ float PH(const float* p, int b, int i, int j)
{ return p[(b << 14) + (i << 7) + j]; }

__device__ __forceinline__ float gY(const float* p, int b, int i, int j)
{
    if (i == 0)   return PH(p, b, 1, j) - PH(p, b, 0, j);
    if (i == 127) return PH(p, b, 127, j) - PH(p, b, 126, j);
    return 0.5f * (PH(p, b, i + 1, j) - PH(p, b, i - 1, j));
}
__device__ __forceinline__ float gX(const float* p, int b, int i, int j)
{
    if (j == 0)   return PH(p, b, i, 1) - PH(p, b, i, 0);
    if (j == 127) return PH(p, b, i, 127) - PH(p, b, i, 126);
    return 0.5f * (PH(p, b, i, j + 1) - PH(p, b, i, j - 1));
}
__device__ __forceinline__ float FXN(const float* p, int b, int i, int j)
{
    const float fx = gX(p, b, i, j), fy = gY(p, b, i, j);
    return fx * rsqrtf(fx * fx + fy * fy + 1e-8f);
}
__device__ __forceinline__ float FYN(const float* p, int b, int i, int j)
{
    const float fx = gX(p, b, i, j), fy = gY(p, b, i, j);
    return fy * rsqrtf(fx * fx + fy * fy + 1e-8f);
}

__global__ void k_cv_reduce(const float* __restrict__ phi, const float* __restrict__ img,
                            float* __restrict__ c1c2)
{
    const int b = blockIdx.x;
    __shared__ float sA[256], sB[256], sC[256], sD[256];
    float s0 = 0.f, s1 = 0.f, s2 = 0.f, s3 = 0.f;
    for (int k = threadIdx.x; k < 16384; k += 256) {
        const float ph = phi[(b << 14) + k];
        const float Hv = 0.5f * (1.f + (2.f / PI_F) * atanf(ph));
        const float im = img[(b << 14) + k];
        s0 += im * Hv; s1 += Hv; s2 += im * (1.f - Hv); s3 += (1.f - Hv);
    }
    sA[threadIdx.x] = s0; sB[threadIdx.x] = s1;
    sC[threadIdx.x] = s2; sD[threadIdx.x] = s3;
    __syncthreads();
    for (int st = 128; st > 0; st >>= 1) {
        if (threadIdx.x < st) {
            sA[threadIdx.x] += sA[threadIdx.x + st];
            sB[threadIdx.x] += sB[threadIdx.x + st];
            sC[threadIdx.x] += sC[threadIdx.x + st];
            sD[threadIdx.x] += sD[threadIdx.x + st];
        }
        __syncthreads();
    }
    if (threadIdx.x == 0) {
        c1c2[b]     = sA[0] / (sB[0] + 1e-8f);
        c1c2[8 + b] = sC[0] / (sD[0] + 1e-8f);
    }
}

__global__ void k_cv_update(const float* __restrict__ phi, const float* __restrict__ img,
                            const float* __restrict__ c1c2, float* __restrict__ out)
{
    const int idx = blockIdx.x * 256 + threadIdx.x;
    if (idx >= MTOT) return;
    const int b = idx >> 14, r = idx & 16383;
    const int i = r >> 7, j = r & 127;

    float dfx;
    if (j == 0)        dfx = FXN(phi, b, i, 1) - FXN(phi, b, i, 0);
    else if (j == 127) dfx = FXN(phi, b, i, 127) - FXN(phi, b, i, 126);
    else               dfx = 0.5f * (FXN(phi, b, i, j + 1) - FXN(phi, b, i, j - 1));
    float dfy;
    if (i == 0)        dfy = FYN(phi, b, 1, j) - FYN(phi, b, 0, j);
    else if (i == 127) dfy = FYN(phi, b, 127, j) - FYN(phi, b, 126, j);
    else               dfy = 0.5f * (FYN(phi, b, i + 1, j) - FYN(phi, b, i - 1, j));

    const float kap = dfx + dfy;
    const float ph = PH(phi, b, i, j);
    const float dirac = (1.f / PI_F) / (1.f + ph * ph);
    const float im = img[idx];
    const float d1 = im - c1c2[b], d2 = im - c1c2[8 + b];
    const float force = dirac * (0.1f * kap - d1 * d1 + d2 * d2);
    out[idx] = ph + 0.1f * force;
}

__global__ void k_clip(const float* __restrict__ phi, float* __restrict__ seg)
{
    const int idx = blockIdx.x * 256 + threadIdx.x;
    if (idx >= MTOT) return;
    seg[idx] = fminf(fmaxf(phi[idx], -1.f), 1.f);
}

// xc = [x1 (256ch f16) | clip(phi)*residual (128ch f16)]
__global__ void k_build_xc(const _Float16* __restrict__ x1, const float* __restrict__ phi,
                           const float* __restrict__ res, _Float16* __restrict__ xc)
{
    size_t idx = (size_t)blockIdx.x * 256 + threadIdx.x;
    if (idx >= (size_t)MTOT * 384) return;
    const size_t m = idx / 384;
    const int c = (int)(idx % 384);
    if (c < 256) {
        xc[idx] = x1[m * 256 + c];
    } else {
        const float s = fminf(fmaxf(phi[m], -1.f), 1.f);
        xc[idx] = (_Float16)(s * res[m * 128 + (c - 256)]);
    }
}

// =============================== host side =================================
extern "C" void kernel_launch(void* const* d_in, const int* in_sizes, int n_in,
                              void* d_out, int out_size, void* d_ws, size_t ws_size,
                              hipStream_t stream)
{
    const float* input    = (const float*)d_in[0];
    const float* residual = (const float*)d_in[1];
    const float* img      = (const float*)d_in[2];
    const float* cr_w0 = (const float*)d_in[3],  *cr_b0 = (const float*)d_in[4];
    const float* cr_g0 = (const float*)d_in[5],  *cr_be0 = (const float*)d_in[6];
    const float* cr_w1 = (const float*)d_in[7],  *cr_b1 = (const float*)d_in[8];
    const float* cr_g1 = (const float*)d_in[9],  *cr_be1 = (const float*)d_in[10];
    const float* at_w0 = (const float*)d_in[11], *at_b0 = (const float*)d_in[12];
    const float* at_g0 = (const float*)d_in[13], *at_be0 = (const float*)d_in[14];
    const float* at_w1 = (const float*)d_in[15], *at_b1 = (const float*)d_in[16];
    const float* at_g1 = (const float*)d_in[17], *at_be1 = (const float*)d_in[18];
    const float* at_w2 = (const float*)d_in[19], *at_b2 = (const float*)d_in[20];
    const float* at_g2 = (const float*)d_in[21], *at_be2 = (const float*)d_in[22];
    const float* at_w3 = (const float*)d_in[23], *at_b3 = (const float*)d_in[24];
    const float* at_g3 = (const float*)d_in[25], *at_be3 = (const float*)d_in[26];
    const float* at_w4 = (const float*)d_in[27], *at_b4 = (const float*)d_in[28];
    const float* at_g4 = (const float*)d_in[29], *at_be4 = (const float*)d_in[30];
    const float* dc_w = (const float*)d_in[31],  *dc_b = (const float*)d_in[32];
    const float* dc_g = (const float*)d_in[33],  *dc_be = (const float*)d_in[34];

    constexpr size_t MB = 1ull << 20;
    char* wsp = (char*)d_ws;
    // big buffers (with lifetime-based reuse)
    _Float16* xf16  = (_Float16*)(wsp + 0);           // 64MB  : concat f16
    float*    rawA  = (float*)  (wsp + 64 * MB);      // 128MB : conv raw f32
    _Float16* hf16  = (_Float16*)(wsp + 192 * MB);    // 64MB
    _Float16* x1f16 = (_Float16*)(wsp + 256 * MB);    // 64MB
    float*    a2bn  = (float*)  (wsp + 320 * MB);     // 64MB
    float*    a1raw = (float*)  (wsp + 64 * MB);      // alias rawA lower half
    float*    a2raw = (float*)  (wsp + 128 * MB);     // alias rawA upper half
    float*    a_act = (float*)  (wsp + 0);            // alias xf16 (dead by then)
    _Float16* xcf16 = (_Float16*)(wsp + 0);           // 96MB, alias [0,96) (dead)
    // small region
    char* sm = wsp + 384 * MB;
    float* araw  = (float*)(sm + 0);
    float* a_sig = (float*)(sm + 512 * 1024);
    float* t3raw = (float*)(sm + 1 * MB);
    float* t4raw = (float*)(sm + 1536 * 1024);
    float* phiA  = (float*)(sm + 2 * MB);
    float* phiB  = (float*)(sm + 2 * MB + 512 * 1024);
    float* imr   = (float*)(sm + 3 * MB);
    float* c1c2  = (float*)(sm + 3584 * 1024);
    float* stats = (float*)(sm + 4 * MB);   // stage s: mean=stats+s*512, inv=+256
    _Float16* wt0   = (_Float16*)(sm + 6 * MB);
    _Float16* wt1   = (_Float16*)(sm + 8 * MB);
    _Float16* wtat0 = (_Float16*)(sm + 10 * MB);
    _Float16* wtat1 = (_Float16*)(sm + 10 * MB + 512 * 1024);
    _Float16* wtdc  = (_Float16*)(sm + 11 * MB);

    float* out    = (float*)d_out;
    float* xd_out = out;                       // [8,256,256,128]
    float* edt_out = out + 67108864;
    float* seg_out = edt_out + 131072;
    float* att_out = seg_out + 131072;

#define STM(s) (stats + (s) * 512)
#define STI(s) (stats + (s) * 512 + 256)

    // --- prep: concat->f16, weights->f16 [tap][Cout][Cin] ---
    k_concat_cvt<<<131072, 256, 0, stream>>>(input, residual, xf16);
    k_wt_cvt<<<2304, 256, 0, stream>>>(cr_w0, wt0, 9, 256, 256);
    k_wt_cvt<<<2304, 256, 0, stream>>>(cr_w1, wt1, 9, 256, 256);
    k_wt_cvt<<<128, 256, 0, stream>>>(at_w0, wtat0, 1, 256, 128);
    k_wt_cvt<<<64, 256, 0, stream>>>(at_w1, wtat1, 1, 128, 128);
    k_wt_cvt<<<768, 256, 0, stream>>>(dc_w, wtdc, 4, 384, 128);

    // --- ConvRelu2: conv0 (3x3, 256->256) ---
    k_gemm_wmma<<<dim3(4, 1024), 256, 0, stream>>>(xf16, 256, wt0, cr_b0, rawA,
                                                   256, 256, 9, 0, 0, 0);
    k_stats<<<256, 256, 0, stream>>>(rawA, MTOT, 256, STM(0), STI(0));
    k_bn_apply<<<131072, 256, 0, stream>>>(rawA, MTOT, 256, STM(0), STI(0),
                                           cr_g0, cr_be0, 1, 0, nullptr, hf16);
    // --- conv1 (3x3, 256->256) ---
    k_gemm_wmma<<<dim3(4, 1024), 256, 0, stream>>>(hf16, 256, wt1, cr_b1, rawA,
                                                   256, 256, 9, 0, 0, 0);
    k_stats<<<256, 256, 0, stream>>>(rawA, MTOT, 256, STM(1), STI(1));
    k_bn_apply<<<131072, 256, 0, stream>>>(rawA, MTOT, 256, STM(1), STI(1),
                                           cr_g1, cr_be1, 1, 0, nullptr, x1f16);

    // --- attention 1x1 GEMMs: a1 = conv(x1, 256->128), a2 = conv(residual, 128->128)
    k_gemm_wmma<<<dim3(2, 1024), 256, 0, stream>>>(x1f16, 256, wtat0, at_b0, a1raw,
                                                   256, 128, 1, 0, 0, 0);
    k_gemm_wmma<<<dim3(2, 1024), 256, 0, stream>>>(xf16 + 128, 256, wtat1, at_b1, a2raw,
                                                   128, 128, 1, 0, 0, 0);
    k_stats<<<128, 256, 0, stream>>>(a1raw, MTOT, 128, STM(2), STI(2));
    k_stats<<<128, 256, 0, stream>>>(a2raw, MTOT, 128, STM(3), STI(3));
    k_bn_apply<<<65536, 256, 0, stream>>>(a2raw, MTOT, 128, STM(3), STI(3),
                                          at_g1, at_be1, 0, 0, a2bn, nullptr);
    k_add_bn_relu<<<65536, 256, 0, stream>>>(a1raw, a2raw,
                                             STM(2), STI(2), at_g0, at_be0,
                                             STM(3), STI(3), at_g1, at_be1, a_act);

    // --- a = sigmoid(BN(conv(a_act, 128->1))) ---
    k_dot128<<<512, 256, 0, stream>>>(a_act, at_w2, at_b2, araw);
    k_stats<<<1, 256, 0, stream>>>(araw, MTOT, 1, STM(4), STI(4));
    k_bn_apply<<<512, 256, 0, stream>>>(araw, MTOT, 1, STM(4), STI(4),
                                        at_g2, at_be2, 0, 1, a_sig, nullptr);

    // --- att_in = sigmoid(BN(w4*(imr + BN(conv(a2,w3))) + b4)) ---
    k_dot128<<<512, 256, 0, stream>>>(a2bn, at_w3, at_b3, t3raw);
    k_stats<<<1, 256, 0, stream>>>(t3raw, MTOT, 1, STM(5), STI(5));
    k_resize_sigmoid<<<512, 256, 0, stream>>>(img, imr);
    k_t4<<<512, 256, 0, stream>>>(t3raw, imr, STM(5), STI(5), at_g3, at_be3,
                                  at_w4, at_b4, t4raw);
    k_stats<<<1, 256, 0, stream>>>(t4raw, MTOT, 1, STM(6), STI(6));
    k_bn_apply<<<512, 256, 0, stream>>>(t4raw, MTOT, 1, STM(6), STI(6),
                                        at_g4, at_be4, 0, 1, att_out, nullptr);

    // --- edt = conv_phi(a); Chan-Vese 30 iterations ---
    k_conv_phi<<<512, 256, 0, stream>>>(a_sig, edt_out, phiA);
    {
        float* pc = phiA; float* pn = phiB;
        for (int it = 0; it < 30; ++it) {
            k_cv_reduce<<<8, 256, 0, stream>>>(pc, att_out, c1c2);
            k_cv_update<<<512, 256, 0, stream>>>(pc, att_out, c1c2, pn);
            float* tmp = pc; pc = pn; pn = tmp;
        }
        // after 30 swaps, result is back in phiA
        k_clip<<<512, 256, 0, stream>>>(pc, seg_out);
        // --- xc = [x1 | seg*residual] f16 ---
        k_build_xc<<<196608, 256, 0, stream>>>(x1f16, pc, residual, xcf16);
    }

    // --- transposed conv 2x2 stride 2: 4 tap-GEMMs scattering into d_out ---
    for (int t = 0; t < 4; ++t) {
        const int ty = t >> 1, tx = t & 1;
        k_gemm_wmma<<<dim3(2, 1024), 256, 0, stream>>>(
            xcf16, 384, wtdc + (size_t)t * 384 * 128, dc_b, xd_out,
            384, 128, 1, 1, ty, tx);
    }
    // --- final BN over [8*256*256, 128], in place on d_out ---
    k_stats<<<128, 256, 0, stream>>>(xd_out, 524288, 128, STM(7), STI(7));
    k_bn_apply<<<262144, 256, 0, stream>>>(xd_out, 524288, 128, STM(7), STI(7),
                                           dc_g, dc_be, 0, 0, xd_out, nullptr);
#undef STM
#undef STI
    (void)in_sizes; (void)n_in; (void)out_size; (void)ws_size;
}
